// EncoderRNN_64733747085812
// MI455X (gfx1250) — compile-verified, matrix operand order checked by reading, not run on hardware
//
#include <hip/hip_runtime.h>
#include <hip/hip_bf16.h>
#include <math.h>

// ---------------------------------------------------------------------------
// Types for CDNA5 WMMA
// ---------------------------------------------------------------------------
typedef __attribute__((ext_vector_type(16))) __bf16 v16bf;
typedef __attribute__((ext_vector_type(8)))  float  v8f;

union Frag16 {
    uint4 u[2];   // 32 bytes
    v16bf v;      // 16 bf16 = 32 bytes
};

__device__ inline v8f vzero8() {
    v8f v;
#pragma unroll
    for (int i = 0; i < 8; ++i) v[i] = 0.0f;
    return v;
}

#define WMMA_BF16(a, b, c) \
    __builtin_amdgcn_wmma_f32_16x16x32_bf16(false, (a), false, (b), (short)0, (c), false, false)

// Async Global->LDS copy of 32 contiguous bytes (two b128 ops; INST_OFFSET
// applies to both the LDS and the global address per the CDNA5 ISA).
__device__ inline void async_copy_32B(unsigned lds_byte_off, const void* gsrc) {
    asm volatile(
        "global_load_async_to_lds_b128 %0, %1, off\n\t"
        "global_load_async_to_lds_b128 %0, %1, off offset:16"
        :: "v"(lds_byte_off), "v"(gsrc) : "memory");
}
__device__ inline void wait_async_le4() { asm volatile("s_wait_asynccnt 0x4" ::: "memory"); }
__device__ inline void wait_async_0()   { asm volatile("s_wait_asynccnt 0x0" ::: "memory"); }

// Problem constants
#define Bn 32
#define Tn 512
#define Dn 1024
#define Un 1024
#define G3 (3 * Un)   // 3072

// ---------------------------------------------------------------------------
// Elementwise convert f32 -> bf16
// ---------------------------------------------------------------------------
__global__ __launch_bounds__(256) void k_f32_to_bf16(const float* __restrict__ in,
                                                     __bf16* __restrict__ out,
                                                     size_t n) {
    size_t i = (size_t)blockIdx.x * blockDim.x + threadIdx.x;
    if (i < n) out[i] = (__bf16)in[i];
}

// Convert + transpose: in is [K][N] f32 row-major, out is [N][K] bf16 row-major
__global__ __launch_bounds__(256) void k_transpose_bf16(const float* __restrict__ in,
                                                        __bf16* __restrict__ out,
                                                        int K, int N) {
    size_t i = (size_t)blockIdx.x * blockDim.x + threadIdx.x;
    size_t total = (size_t)K * N;
    if (i < total) {
        int n = (int)(i / K);
        int k = (int)(i % K);
        out[i] = (__bf16)in[(size_t)k * N + n];
    }
}

// Zero a region (16B granules)
__global__ __launch_bounds__(256) void k_zero(uint4* __restrict__ p, size_t n16) {
    size_t i = (size_t)blockIdx.x * blockDim.x + threadIdx.x;
    if (i < n16) { uint4 z; z.x = z.y = z.z = z.w = 0u; p[i] = z; }
}

// ---------------------------------------------------------------------------
// Big projection GEMM: C[M][N] = A[M][K] (bf16) x Bt[N][K] (bf16, pre-transposed)
//                      + bias[N], f32 accumulate/output.
// Block tile 128x128x32, 8 waves (2x4), each wave: 64x32 via 4x2 accum tiles.
// Tiles staged with double-buffered GLOBAL_LOAD_ASYNC_TO_LDS_B128 (ASYNCcnt).
// ---------------------------------------------------------------------------
#define GBM 128
#define GBN 128
#define GBK 32
#define ATILE (GBM * GBK)   // elements per A buffer (4096 bf16 = 8KB)
#define BTILE (GBN * GBK)

__global__ __launch_bounds__(256) void k_gemm_bf16(const __bf16* __restrict__ A,
                                                   const __bf16* __restrict__ Bt,
                                                   const float* __restrict__ bias,
                                                   float* __restrict__ C,
                                                   int M, int N, int K) {
    // [bufA0 | bufA1 | bufB0 | bufB1]
    __shared__ __bf16 smem[2 * ATILE + 2 * BTILE];

    const int tid  = threadIdx.x;
    const int lane = tid & 31;
    const int wave = tid >> 5;
    const int bm = blockIdx.x * GBM;
    const int bn = blockIdx.y * GBN;

    const int wm = wave & 1;        // 0..1
    const int wn = wave >> 1;       // 0..3
    const int wMbase = wm * 64;     // 4 m-tiles of 16
    const int wNbase = wn * 32;     // 2 n-tiles of 16

    const int m = lane & 15;        // row/col within 16-wide tile
    const int g = lane >> 4;        // half-select for K groups

    v8f acc[4][2];
#pragma unroll
    for (int i = 0; i < 4; ++i)
#pragma unroll
        for (int j = 0; j < 2; ++j) acc[i][j] = vzero8();

    // Staging geometry: 256 threads, each copies one 32B run of one row.
    const int srow  = tid >> 1;          // 0..127
    const int scolh = (tid & 1) * 16;    // element offset 0 or 16

    const unsigned smem_base = (unsigned)(uintptr_t)(&smem[0]);
    const unsigned ldsA[2] = { smem_base + (unsigned)((0 * ATILE + srow * GBK + scolh) * 2),
                               smem_base + (unsigned)((1 * ATILE + srow * GBK + scolh) * 2) };
    const unsigned ldsB[2] = { smem_base + (unsigned)((2 * ATILE + 0 * BTILE + srow * GBK + scolh) * 2),
                               smem_base + (unsigned)((2 * ATILE + 1 * BTILE + srow * GBK + scolh) * 2) };
    const __bf16* gA = A  + (size_t)(bm + srow) * K + scolh;
    const __bf16* gB = Bt + (size_t)(bn + srow) * K + scolh;

    const int NK = K / GBK;

    // Prologue: stage chunk 0 into buffer 0
    async_copy_32B(ldsA[0], gA);
    async_copy_32B(ldsB[0], gB);

    for (int i = 0; i < NK; ++i) {
        const int buf = i & 1;
        // Issue next chunk into the other buffer
        if (i + 1 < NK) {
            const int nb = (i + 1) & 1;
            async_copy_32B(ldsA[nb], gA + (size_t)(i + 1) * GBK);
            async_copy_32B(ldsB[nb], gB + (size_t)(i + 1) * GBK);
            wait_async_le4();   // chunk i complete (async loads retire in order)
        } else {
            wait_async_0();
        }
        __syncthreads();        // chunk i visible to all waves

        const __bf16* sA = smem + buf * ATILE;
        const __bf16* sB = smem + 2 * ATILE + buf * BTILE;

        Frag16 fa[4], fb[2];
#pragma unroll
        for (int mt = 0; mt < 4; ++mt) {
            const uint4* p = (const uint4*)(sA + (wMbase + mt * 16 + m) * GBK + g * 8);
            fa[mt].u[0] = p[0];   // K = g*8 .. g*8+7
            fa[mt].u[1] = p[2];   // K = 16+g*8 .. 16+g*8+7
        }
#pragma unroll
        for (int nt = 0; nt < 2; ++nt) {
            const uint4* p = (const uint4*)(sB + (wNbase + nt * 16 + m) * GBK + g * 8);
            fb[nt].u[0] = p[0];
            fb[nt].u[1] = p[2];
        }
#pragma unroll
        for (int mt = 0; mt < 4; ++mt)
#pragma unroll
            for (int nt = 0; nt < 2; ++nt)
                acc[mt][nt] = WMMA_BF16(fa[mt].v, fb[nt].v, acc[mt][nt]);

        __syncthreads();        // all waves done reading buf i (frees buf for chunk i+2)
    }

    // Epilogue: C layout — lane 0-15: row = r (+mtile), col = lane; lane 16-31: row = r+8
#pragma unroll
    for (int nt = 0; nt < 2; ++nt) {
        const int col = bn + wNbase + nt * 16 + m;
        const float bv = bias ? bias[col] : 0.0f;
#pragma unroll
        for (int mt = 0; mt < 4; ++mt) {
#pragma unroll
            for (int r = 0; r < 8; ++r) {
                const int row = bm + wMbase + mt * 16 + r + g * 8;
                C[(size_t)row * N + col] = acc[mt][nt][r] + bv;
            }
        }
    }
}

// ---------------------------------------------------------------------------
// Persistent GRU recurrence. One launch handles one layer, both directions.
// Grid = 32 blocks x 128 threads; 16 blocks per direction. Each wave owns a
// 16-column tile of h_new and computes three gate column tiles (z, r, h).
// Per-direction step barrier: CDNA5 cluster barrier when the direction's 16
// workgroups are launched as one cluster, else monotonic atomic counter.
// ---------------------------------------------------------------------------
__global__ __launch_bounds__(128) void k_gru_recur(const float* __restrict__ xp_f,
                                                   const float* __restrict__ xp_b,
                                                   const __bf16* __restrict__ Ut_f,
                                                   const __bf16* __restrict__ Ut_b,
                                                   const float* __restrict__ bias_f,
                                                   const float* __restrict__ bias_b,
                                                   __bf16* __restrict__ hbf,   // [2dir][2buf][B][U]
                                                   float*  __restrict__ hf32,  // [2dir][2buf][B][U]
                                                   __bf16* __restrict__ out_bf,  // layer0: [T][B][2U] or null
                                                   float*  __restrict__ out_f32, // layer1: [B][T][2U] or null
                                                   float*  __restrict__ states,  // layer1: [2][B][U] or null
                                                   unsigned* __restrict__ barrier_cnt, // [2]
                                                   int nblocks) {
    const int half = nblocks >> 1;
    const int dir  = (blockIdx.x >= half) ? 1 : 0;
    const int wb   = blockIdx.x - dir * half;
    const int wave = threadIdx.x >> 5;
    const int lane = threadIdx.x & 31;
    const int ctile = wb * 4 + wave;     // 0..63
    const int c0 = ctile * 16;

    const float*  xp   = dir ? xp_b   : xp_f;
    const __bf16* Ut   = dir ? Ut_b   : Ut_f;
    const float*  bias = dir ? bias_b : bias_f;
    unsigned*     cnt  = barrier_cnt + dir;
    const bool in_cluster = (__builtin_amdgcn_cluster_id_x() > 0);

    const int m = lane & 15;
    const int g = lane >> 4;
    const int col = c0 + m;

    const float bz = bias[col];
    const float br = bias[Un + col];
    const float bh = bias[2 * Un + col];

    for (int s = 0; s < Tn; ++s) {
        const int t   = dir ? (Tn - 1 - s) : s;
        const int cur = s & 1;
        const int nxt = cur ^ 1;

        const __bf16* hp = hbf + ((size_t)dir * 2 + cur) * (size_t)Bn * Un;

        v8f acc[3][2];
#pragma unroll
        for (int ga = 0; ga < 3; ++ga)
#pragma unroll
            for (int mt = 0; mt < 2; ++mt) acc[ga][mt] = vzero8();

        for (int k0 = 0; k0 < Un; k0 += 32) {
            Frag16 fa[2], fb[3];
#pragma unroll
            for (int mt = 0; mt < 2; ++mt) {
                const uint4* p = (const uint4*)(hp + (size_t)(mt * 16 + m) * Un + k0 + g * 8);
                fa[mt].u[0] = p[0];
                fa[mt].u[1] = p[2];
            }
#pragma unroll
            for (int ga = 0; ga < 3; ++ga) {
                const uint4* p = (const uint4*)(Ut + (size_t)(ga * Un + c0 + m) * Un + k0 + g * 8);
                fb[ga].u[0] = p[0];
                fb[ga].u[1] = p[2];
            }
#pragma unroll
            for (int ga = 0; ga < 3; ++ga)
#pragma unroll
                for (int mt = 0; mt < 2; ++mt)
                    acc[ga][mt] = WMMA_BF16(fa[mt].v, fb[ga].v, acc[ga][mt]);
        }

        // Gate math + state update
        const float* hpf = hf32 + ((size_t)dir * 2 + cur) * (size_t)Bn * Un;
        float*       hnf = hf32 + ((size_t)dir * 2 + nxt) * (size_t)Bn * Un;
        __bf16*      hnb = hbf  + ((size_t)dir * 2 + nxt) * (size_t)Bn * Un;
        const float* xpt = xp + (size_t)t * Bn * G3;

#pragma unroll
        for (int mt = 0; mt < 2; ++mt) {
#pragma unroll
            for (int r = 0; r < 8; ++r) {
                const int row = mt * 16 + r + g * 8;   // batch index 0..31
                const float* xrow = xpt + (size_t)row * G3;
                const float xz = xrow[col];
                const float xr = xrow[Un + col];
                const float xh = xrow[2 * Un + col];
                const float rz = acc[0][mt][r] + bz;
                const float rr = acc[1][mt][r] + br;
                const float rh = acc[2][mt][r] + bh;
                const float z  = 1.0f / (1.0f + __expf(-(xz + rz)));
                const float rg = 1.0f / (1.0f + __expf(-(xr + rr)));
                const float hh = tanhf(xh + rg * rh);
                const float hprev = hpf[(size_t)row * Un + col];
                const float hn = z * hprev + (1.0f - z) * hh;

                hnf[(size_t)row * Un + col] = hn;
                hnb[(size_t)row * Un + col] = (__bf16)hn;

                if (out_bf)
                    out_bf[((size_t)t * Bn + row) * (2 * Un) + (size_t)dir * Un + col] = (__bf16)hn;
                if (out_f32) {
                    out_f32[((size_t)row * Tn + t) * (2 * Un) + (size_t)dir * Un + col] = hn;
                    if (s == Tn - 1 && states)
                        states[((size_t)dir * Bn + row) * Un + col] = hn;
                }
            }
        }

        // Per-direction step barrier
        __syncthreads();
        if (in_cluster) {
            // 16 workgroups of this direction form one cluster: one wave
            // signals + waits the HW cluster barrier, workgroup barriers
            // bracket it for the other waves.
            if (wave == 0) __builtin_amdgcn_s_cluster_barrier();
        } else {
            if (threadIdx.x == 0) {
                __threadfence();
                atomicAdd(cnt, 1u);
                const unsigned target = (unsigned)half * (unsigned)(s + 1);
                while (__hip_atomic_load(cnt, __ATOMIC_ACQUIRE, __HIP_MEMORY_SCOPE_AGENT) < target) {
                    __builtin_amdgcn_s_sleep(1);
                }
            }
        }
        __syncthreads();
    }
}

// ---------------------------------------------------------------------------
// Host-side orchestration
// ---------------------------------------------------------------------------
extern "C" void kernel_launch(void* const* d_in, const int* in_sizes, int n_in,
                              void* d_out, int out_size, void* d_ws, size_t ws_size,
                              hipStream_t stream) {
    (void)in_sizes; (void)n_in; (void)out_size; (void)ws_size;

    const float* x   = (const float*)d_in[0];
    const float* Wf0 = (const float*)d_in[1];
    const float* Uf0 = (const float*)d_in[2];
    const float* bf0 = (const float*)d_in[3];
    const float* Wb0 = (const float*)d_in[4];
    const float* Ub0 = (const float*)d_in[5];
    const float* bb0 = (const float*)d_in[6];
    const float* Wf1 = (const float*)d_in[7];
    const float* Uf1 = (const float*)d_in[8];
    const float* bf1 = (const float*)d_in[9];
    const float* Wb1 = (const float*)d_in[10];
    const float* Ub1 = (const float*)d_in[11];
    const float* bb1 = (const float*)d_in[12];

    float* out = (float*)d_out;
    float* states = out + (size_t)Bn * Tn * 2 * Un;

    // Workspace layout
    char* ws = (char*)d_ws;
    size_t off = 0;
    auto alloc = [&](size_t bytes) -> size_t {
        bytes = (bytes + 255) & ~(size_t)255;
        size_t o = off; off += bytes; return o;
    };
    const size_t MROWS = (size_t)Bn * Tn;                 // 16384
    size_t o_xbf  = alloc(MROWS * Dn * 2);                // x bf16 [B*T][D]
    size_t o_Wf0t = alloc((size_t)G3 * Dn * 2);           // [3U][D] bf16
    size_t o_Uf0t = alloc((size_t)G3 * Un * 2);
    size_t o_Wb0t = alloc((size_t)G3 * Dn * 2);
    size_t o_Ub0t = alloc((size_t)G3 * Un * 2);
    size_t o_Wf1t = alloc((size_t)G3 * (2 * Un) * 2);     // [3U][2U] bf16
    size_t o_Uf1t = alloc((size_t)G3 * Un * 2);
    size_t o_Wb1t = alloc((size_t)G3 * (2 * Un) * 2);
    size_t o_Ub1t = alloc((size_t)G3 * Un * 2);
    size_t o_xpA  = alloc(MROWS * G3 * 4);                // xp fwd (reused per layer)
    size_t o_xpB  = alloc(MROWS * G3 * 4);                // xp bwd (reused per layer)
    size_t o_y0bf = alloc((size_t)Tn * Bn * 2 * Un * 2);  // layer0 concat output bf16
    size_t o_hbf  = alloc((size_t)2 * 2 * Bn * Un * 2);
    size_t o_hf32 = alloc((size_t)2 * 2 * Bn * Un * 4);
    size_t o_cnt  = alloc(256);

    __bf16* xbf  = (__bf16*)(ws + o_xbf);
    __bf16* Wf0t = (__bf16*)(ws + o_Wf0t);
    __bf16* Uf0t = (__bf16*)(ws + o_Uf0t);
    __bf16* Wb0t = (__bf16*)(ws + o_Wb0t);
    __bf16* Ub0t = (__bf16*)(ws + o_Ub0t);
    __bf16* Wf1t = (__bf16*)(ws + o_Wf1t);
    __bf16* Uf1t = (__bf16*)(ws + o_Uf1t);
    __bf16* Wb1t = (__bf16*)(ws + o_Wb1t);
    __bf16* Ub1t = (__bf16*)(ws + o_Ub1t);
    float*  xpA  = (float*)(ws + o_xpA);
    float*  xpB  = (float*)(ws + o_xpB);
    __bf16* y0bf = (__bf16*)(ws + o_y0bf);
    __bf16* hbf  = (__bf16*)(ws + o_hbf);
    float*  hf32 = (float*)(ws + o_hf32);
    unsigned* cnt = (unsigned*)(ws + o_cnt);

    // --- 1. Convert x to bf16 ---
    {
        size_t n = MROWS * Dn;
        k_f32_to_bf16<<<(unsigned)((n + 255) / 256), 256, 0, stream>>>(x, xbf, n);
    }
    // --- 2. Convert+transpose all weights to bf16 [N][K] ---
    auto launch_tr = [&](const float* in, __bf16* o, int K, int N) {
        size_t n = (size_t)K * N;
        k_transpose_bf16<<<(unsigned)((n + 255) / 256), 256, 0, stream>>>(in, o, K, N);
    };
    launch_tr(Wf0, Wf0t, Dn, G3);
    launch_tr(Uf0, Uf0t, Un, G3);
    launch_tr(Wb0, Wb0t, Dn, G3);
    launch_tr(Ub0, Ub0t, Un, G3);
    launch_tr(Wf1, Wf1t, 2 * Un, G3);
    launch_tr(Uf1, Uf1t, Un, G3);
    launch_tr(Wb1, Wb1t, 2 * Un, G3);
    launch_tr(Ub1, Ub1t, Un, G3);

    dim3 ggrid((unsigned)(MROWS / GBM), (unsigned)(G3 / GBN));

    // --- 3. Layer 0 input projections: xp = x @ W + b[0] ---
    k_gemm_bf16<<<ggrid, 256, 0, stream>>>(xbf, Wf0t, bf0, xpA, (int)MROWS, G3, Dn);
    k_gemm_bf16<<<ggrid, 256, 0, stream>>>(xbf, Wb0t, bb0, xpB, (int)MROWS, G3, Dn);

    const size_t zbytes = ((((size_t)2 * 2 * Bn * Un * 2) + 255) & ~(size_t)255)
                        + ((((size_t)2 * 2 * Bn * Un * 4) + 255) & ~(size_t)255) + 256;
    const size_t zn16 = zbytes / 16;

    // --- 4. Zero h state + barrier counters, then layer 0 recurrence ---
    k_zero<<<(unsigned)((zn16 + 255) / 256), 256, 0, stream>>>((uint4*)(ws + o_hbf), zn16);
    k_gru_recur<<<32, 128, 0, stream>>>(xpA, xpB, Uf0t, Ub0t,
                                        bf0 + G3, bb0 + G3,
                                        hbf, hf32,
                                        y0bf, (float*)nullptr, (float*)nullptr,
                                        cnt, 32);

    // --- 5. Layer 1 input projections: xp = y0 @ W + b[0] (K = 2U) ---
    k_gemm_bf16<<<ggrid, 256, 0, stream>>>(y0bf, Wf1t, bf1, xpA, (int)MROWS, G3, 2 * Un);
    k_gemm_bf16<<<ggrid, 256, 0, stream>>>(y0bf, Wb1t, bb1, xpB, (int)MROWS, G3, 2 * Un);

    // --- 6. Re-zero state/counters, layer 1 recurrence (writes d_out) ---
    k_zero<<<(unsigned)((zn16 + 255) / 256), 256, 0, stream>>>((uint4*)(ws + o_hbf), zn16);
    k_gru_recur<<<32, 128, 0, stream>>>(xpA, xpB, Uf1t, Ub1t,
                                        bf1 + G3, bb1 + G3,
                                        hbf, hf32,
                                        (__bf16*)nullptr, out, states,
                                        cnt, 32);
}
